// ProbabilisticAutoma_19344532701881
// MI455X (gfx1250) — matrix operand order repaired and checked by reading, not compiled, not run on hardware
//
#include <hip/hip_runtime.h>
#include <hip/hip_bf16.h>

typedef _Float16 v16h __attribute__((ext_vector_type(16)));
typedef _Float16 h8   __attribute__((ext_vector_type(8)));
typedef float    v8f  __attribute__((ext_vector_type(8)));

#define B_ 256
#define T_ 512
#define A_ 16
#define S_ 256
#define R_ 4
#define K_ (A_*S_)      /* 4096 contraction dim: k = a*256 + s          */
#define SSTR 264        /* LDS state row stride (f16): 4-bank rotation   */

// ---------------------------------------------------------------------------
// Pre-pass 1: BigB[sp][a*256+s] = f16(trans[a][s][sp])   (rows 0..255)
// Transposed so each WMMA B-fragment is a contiguous 32B lane load.
// ---------------------------------------------------------------------------
__global__ void pa_build_wt(const float* __restrict__ trans,
                            _Float16* __restrict__ bigb) {
    int idx = blockIdx.x * blockDim.x + threadIdx.x;    // = a*65536 + s*256 + sp
    if (idx >= A_ * S_ * S_) return;
    int sp = idx & 255;
    int s  = (idx >> 8) & 255;
    int a  = idx >> 16;
    bigb[(size_t)sp * K_ + a * S_ + s] = (_Float16)trans[idx];
}

// ---------------------------------------------------------------------------
// Pre-pass 2: BigB[256+j][k] = f16( sum_sp trans[a][s][sp] * rew[sp][j] )
// for j<4, zero for j=4..15.  Rewards then come out of the same GEMM:
// pred_rew = Abig @ (Wbig @ rew).
// ---------------------------------------------------------------------------
__global__ void pa_build_wr(const float* __restrict__ trans,
                            const float* __restrict__ rew,
                            _Float16* __restrict__ bigb) {
    int idx = blockIdx.x * blockDim.x + threadIdx.x;    // = k*16 + j
    if (idx >= K_ * 16) return;
    int j = idx & 15;
    int k = idx >> 4;
    float acc = 0.f;
    if (j < R_) {
        int a = k >> 8, s = k & 255;
        const float* tr = trans + ((size_t)a * S_ + s) * S_;
        for (int sp = 0; sp < S_; ++sp) acc += tr[sp] * rew[sp * R_ + j];
    }
    bigb[(size_t)(S_ + j) * K_ + k] = (_Float16)acc;
}

// ---------------------------------------------------------------------------
// Main recurrence kernel.
//   grid  = 16 workgroups, one per 16-batch tile
//   block = 544 threads = 17 waves (wave32):
//     waves 0..15 : s' column tiles 16w..16w+15  -> next state + pred_states
//     wave  16    : reward tile (WR columns)     -> pred_rew
// Per step: one [16 x 4096] x [4096 x 260] WMMA GEMM, with the A-matrix
// fragment built on the fly as action[b,a] * stateH fragment (v_pk_mul_f16).
// B stream (2.13 MB/step, L2-resident) is prefetched 2 a-blocks ahead.
// ---------------------------------------------------------------------------
__global__ __launch_bounds__(544) void pa_recurrence(
    const float*    __restrict__ action,     // [B][T][A]
    const _Float16* __restrict__ bigb,       // [272][4096]
    float*          __restrict__ out_states, // [B][T][S]
    float*          __restrict__ out_rew)    // [B][T][R]
{
    __shared__ _Float16 stateH[16 * SSTR];   // current state, f16
    __shared__ float    actLDS[16 * A_];     // action block for current t

    const int tid  = threadIdx.x;
    const int wave = tid >> 5;
    const int lane = tid & 31;
    const int row  = lane & 15;   // A-matrix M row (batch within tile)
    const int half = lane >> 4;   // 0: K low half, 1: K high half
    const int b0   = blockIdx.x * 16;

    // init state = one-hot on state 0
    for (int i = tid; i < 16 * SSTR; i += 544) stateH[i] = (_Float16)0.f;
    __syncthreads();
    if (tid < 16) stateH[tid * SSTR] = (_Float16)1.f;

    // Per-lane B base: lane's N column row of BigB; +half*16 selects K+16 half
    const int brow = (wave < 16) ? (wave * 16 + row) : (S_ + row);
    const _Float16* bbase = bigb + (size_t)brow * K_ + half * 16;

    // action block for t=0
    if (tid < 256)
        actLDS[tid] = action[((size_t)(b0 + (tid >> 4)) * T_) * A_ + (tid & 15)];
    __syncthreads();

    for (int t = 0; t < T_; ++t) {
        // warm the first two a-blocks of this step's B stream
        #pragma unroll
        for (int pf = 0; pf < 8; ++pf)
            __builtin_prefetch(bbase + pf * 64, 0, 3);

        v8f acc = {};
        const _Float16* sb = &stateH[row * SSTR + half * 16];
        #pragma unroll 2
        for (int a = 0; a < A_; ++a) {
            _Float16 ah = (_Float16)actLDS[row * A_ + a];
            const _Float16* bb = bbase + a * S_;
            if (a < A_ - 2) {
                // prefetch the a+2 block (512B/lane, 4 cachelines) of B stream
                __builtin_prefetch(bb + 2 * S_,       0, 3);
                __builtin_prefetch(bb + 2 * S_ + 64,  0, 3);
                __builtin_prefetch(bb + 2 * S_ + 128, 0, 3);
                __builtin_prefetch(bb + 2 * S_ + 192, 0, 3);
            }
            #pragma unroll
            for (int sf = 0; sf < 8; ++sf) {
                // A-fragment: lanes 0-15 take K[0:16] of the 32-chunk,
                // lanes 16-31 take K[16:32]  (contiguous, matches B)
                h8 lo = *(const h8*)(sb + sf * 32);
                h8 hi = *(const h8*)(sb + sf * 32 + 8);
                v16h af = __builtin_shufflevector(lo, hi,
                          0,1,2,3,4,5,6,7,8,9,10,11,12,13,14,15);
                af = af * ah;                  // fold action weight into A
                v16h bf = *(const v16h*)(bb + sf * 32);
                acc = __builtin_amdgcn_wmma_f32_16x16x32_f16(
                          false, af, false, bf, (short)0, acc, false, false);
            }
        }
        __syncthreads();                       // all reads of stateH/actLDS done

        if (wave < 16) {
            const int sp = wave * 16 + row;    // this lane's s' column
            #pragma unroll
            for (int v = 0; v < 8; ++v) {      // C layout: M = v + 8*half
                int bl = v + half * 8;
                out_states[((size_t)(b0 + bl) * T_ + t) * S_ + sp] = acc[v];
                stateH[bl * SSTR + sp] = (_Float16)acc[v];
            }
        } else if (row < R_) {
            #pragma unroll
            for (int v = 0; v < 8; ++v) {
                int bl = v + half * 8;
                out_rew[((size_t)(b0 + bl) * T_ + t) * R_ + row] = acc[v];
            }
        }

        // action block for t+1 (all reads of actLDS[t] fenced by barrier above)
        if (t + 1 < T_ && tid < 256)
            actLDS[tid] =
                action[((size_t)(b0 + (tid >> 4)) * T_ + (t + 1)) * A_ + (tid & 15)];
        __syncthreads();                       // new stateH + actLDS visible
    }
}

// ---------------------------------------------------------------------------
extern "C" void kernel_launch(void* const* d_in, const int* in_sizes, int n_in,
                              void* d_out, int out_size, void* d_ws, size_t ws_size,
                              hipStream_t stream) {
    const float* action = (const float*)d_in[0];   // [256][512][16]
    const float* trans  = (const float*)d_in[1];   // [16][256][256]
    const float* rew    = (const float*)d_in[2];   // [256][4]

    float* out_states = (float*)d_out;                              // [B][T][S]
    float* out_rew    = (float*)d_out + (size_t)B_ * T_ * S_;       // [B][T][R]

    _Float16* bigb = (_Float16*)d_ws;   // 272 * 4096 * 2B = ~2.13 MB

    pa_build_wt<<<(A_ * S_ * S_ + 255) / 256, 256, 0, stream>>>(trans, bigb);
    pa_build_wr<<<(K_ * 16 + 255) / 256, 256, 0, stream>>>(trans, rew, bigb);
    pa_recurrence<<<B_ / 16, 544, 0, stream>>>(action, bigb, out_states, out_rew);
}
